// VariationalGCNEncoder_20272245637270
// MI455X (gfx1250) — matrix operand order repaired and checked by reading, not compiled, not run on hardware
//
#include <hip/hip_runtime.h>

// ---------------------------------------------------------------------------
// Variational GCN encoder on gfx1250 (MI455X).
//  - GEMMs via V_WMMA_F32_16X16X4_F32 (fallback: v_wmma_f32_16x16x32_f16)
//  - edge aggregation via global_atomic_add_f32 (L2-resident: H fits in 192MB L2)
//  - layer 2+3 fused into one [N,128]x[128,128] GEMM with concatenated weights
// ---------------------------------------------------------------------------

typedef __attribute__((ext_vector_type(2)))  float    v2f;
typedef __attribute__((ext_vector_type(8)))  float    v8f;
typedef __attribute__((ext_vector_type(16))) _Float16 v16h;

#if __has_builtin(__builtin_amdgcn_wmma_f32_16x16x4_f32)
#define USE_F32_WMMA 1
#else
#define USE_F32_WMMA 0
#endif

#define CH 128   // channel width of both GEMMs (HID and fused [mu|logstd])
#define KC 64    // K-chunk of B staged in LDS (32KB)

// ---------------------------------------------------------------------------
// degree / normalization
// ---------------------------------------------------------------------------
__global__ void vgcn_deg_init(float* __restrict__ deg, int n) {
  int i = blockIdx.x * blockDim.x + threadIdx.x;
  if (i < n) deg[i] = 1.0f;                       // self-loop contribution
}

__global__ void vgcn_deg_accum(const int* __restrict__ dst, float* __restrict__ deg, int E) {
  int e = blockIdx.x * blockDim.x + threadIdx.x;
  if (e < E) atomicAdd(&deg[dst[e]], 1.0f);
}

__global__ void vgcn_dinv(float* __restrict__ deg, int n) {
  int i = blockIdx.x * blockDim.x + threadIdx.x;
  if (i < n) deg[i] = __frsqrt_rn(deg[i]);        // deg >= 1 always (self loops)
}

// ---------------------------------------------------------------------------
// GEMM: C[M,128] = A[M,128] * B[128,128], all f32 row-major, M % 16 == 0.
// One block = 16-row strip; 8 waves, one 16x16 tile each. WMMA per ISA layout.
// ---------------------------------------------------------------------------
__global__ __launch_bounds__(256) void vgcn_gemm128(const float* __restrict__ A,
                                                    const float* __restrict__ B,
                                                    float* __restrict__ C, int M) {
  __shared__ float sA[16 * CH];   // 8 KB: A strip for this block
  __shared__ float sB[KC * CH];   // 32 KB: K-chunk of B
  const int t    = threadIdx.x;
  const int row0 = blockIdx.x * 16;

  for (int i = t; i < 16 * CH; i += 256)
    sA[i] = A[(size_t)row0 * CH + i];

  const int wave = t >> 5;
  const int lane = t & 31;
  const int tm   = lane & 15;     // M index (A) / N index (B, C/D)
  const int th   = lane >> 4;     // lane-half selects K group / upper M rows
  const int col0 = wave * 16;

  v8f acc = {};
  for (int kk = 0; kk < CH; kk += KC) {
    __syncthreads();
    for (int i = t; i < KC * CH; i += 256)
      sB[i] = B[kk * CH + i];
    __syncthreads();

#if USE_F32_WMMA
    #pragma unroll
    for (int k = 0; k < KC; k += 4) {
      // A 16x4 f32: lane=M, v0/v1 hold K = 2*th + {0,1}  (ISA 7.12.2)
      v2f a, b;
      a.x = sA[tm * CH + kk + k + 2 * th + 0];
      a.y = sA[tm * CH + kk + k + 2 * th + 1];
      // B 4x16 f32: lane=N, rows striped across VGPRs/lane-halves (mirror of A)
      b.x = sB[(k + 2 * th + 0) * CH + col0 + tm];
      b.y = sB[(k + 2 * th + 1) * CH + col0 + tm];
      acc = __builtin_amdgcn_wmma_f32_16x16x4_f32(false, a, false, b,
                                                  (short)0, acc, false, false);
    }
#else
    #pragma unroll
    for (int k = 0; k < KC; k += 32) {
      // f16 fallback: A 16-bit 16x32 layout per ISA 7.12.2
      v16h a, b;
      #pragma unroll
      for (int j = 0; j < 16; ++j) {
        int i  = j >> 1;
        int ka = ((i & 4) ? 16 : 0) + 8 * th + 2 * (i & 3) + (j & 1);
        a[j] = (_Float16)sA[tm * CH + kk + k + ka];
        int kb = ((i & 4) ? 16 : 0) + 8 * th + 2 * (i & 3) + (j & 1);
        b[j] = (_Float16)sB[(k + kb) * CH + col0 + tm];
      }
      acc = __builtin_amdgcn_wmma_f32_16x16x32_f16(false, a, false, b,
                                                   (short)0, acc, false, false);
    }
#endif
  }

  // C/D 16x16 f32: VGPR r -> row r (lanes 0-15) / row r+8 (lanes 16-31)
  #pragma unroll
  for (int r = 0; r < 8; ++r)
    C[(size_t)(row0 + r + 8 * th) * CH + col0 + tm] = acc[r];
}

// ---------------------------------------------------------------------------
// aggregation (layer 1): agg[i] = dinv[i]^2 * H[i] + b ; then += over edges
// ---------------------------------------------------------------------------
__global__ __launch_bounds__(128) void vgcn_agg_init(const float* __restrict__ H,
                                                     const float* __restrict__ dinv,
                                                     const float* __restrict__ bias,
                                                     float* __restrict__ agg) {
  int i = blockIdx.x, c = threadIdx.x;
  float d = dinv[i];
  agg[(size_t)i * CH + c] = d * d * H[(size_t)i * CH + c] + bias[c];
}

__global__ __launch_bounds__(256) void vgcn_agg_edges(const float* __restrict__ H,
                                                      const int* __restrict__ src,
                                                      const int* __restrict__ dst,
                                                      const float* __restrict__ dinv,
                                                      float* __restrict__ agg, int E) {
  int e = blockIdx.x * 2 + (threadIdx.x >> 7);
  int c = threadIdx.x & 127;
  if (e < E) {
    int s = src[e], d = dst[e];
    float coef = dinv[s] * dinv[d];
    atomicAdd(&agg[(size_t)d * CH + c], coef * H[(size_t)s * CH + c]);
  }
}

// relu -> row L2 normalize (second relu is a no-op on non-negative data)
__global__ __launch_bounds__(128) void vgcn_relu_norm(const float* __restrict__ agg,
                                                      float* __restrict__ H) {
  __shared__ float red[128];
  int i = blockIdx.x, c = threadIdx.x;
  float v = fmaxf(agg[(size_t)i * CH + c], 0.0f);
  red[c] = v * v;
  __syncthreads();
  #pragma unroll
  for (int s = 64; s > 0; s >>= 1) {
    if (c < s) red[c] += red[c + s];
    __syncthreads();
  }
  float nrm = sqrtf(red[0]);
  H[(size_t)i * CH + c] = v / fmaxf(nrm, 1e-12f);
}

// ---------------------------------------------------------------------------
// fused layer 2/3: Wcat = [Wmu | Wls] (128x128)
// ---------------------------------------------------------------------------
__global__ __launch_bounds__(128) void vgcn_wcat(const float* __restrict__ Wmu,
                                                 const float* __restrict__ Wls,
                                                 float* __restrict__ Wcat) {
  int k = blockIdx.x, c = threadIdx.x;
  Wcat[k * CH + c] = (c < 64) ? Wmu[k * 64 + c] : Wls[k * 64 + (c - 64)];
}

__device__ __forceinline__ int out_addr(int n, int i, int c) {
  // c<64 -> mu[i,c] ; c>=64 -> logstd[i,c-64] (tuple concatenated flat)
  return (c < 64) ? (i * 64 + c) : ((n + i) * 64 + (c - 64));
}

__global__ __launch_bounds__(128) void vgcn_agg_init_out(const float* __restrict__ G,
                                                         const float* __restrict__ dinv,
                                                         const float* __restrict__ bmu,
                                                         const float* __restrict__ bls,
                                                         float* __restrict__ out, int n) {
  int i = blockIdx.x, c = threadIdx.x;
  float d = dinv[i];
  float b = (c < 64) ? bmu[c] : bls[c - 64];
  out[out_addr(n, i, c)] = d * d * G[(size_t)i * CH + c] + b;
}

__global__ __launch_bounds__(256) void vgcn_agg_edges_out(const float* __restrict__ G,
                                                          const int* __restrict__ src,
                                                          const int* __restrict__ dst,
                                                          const float* __restrict__ dinv,
                                                          float* __restrict__ out,
                                                          int n, int E) {
  int e = blockIdx.x * 2 + (threadIdx.x >> 7);
  int c = threadIdx.x & 127;
  if (e < E) {
    int s = src[e], d = dst[e];
    float coef = dinv[s] * dinv[d];
    atomicAdd(&out[out_addr(n, d, c)], coef * G[(size_t)s * CH + c]);
  }
}

// ---------------------------------------------------------------------------
extern "C" void kernel_launch(void* const* d_in, const int* in_sizes, int n_in,
                              void* d_out, int out_size, void* d_ws, size_t ws_size,
                              hipStream_t stream) {
  const float* x   = (const float*)d_in[0];
  const int*   ei  = (const int*)  d_in[1];
  const float* W1  = (const float*)d_in[2];
  const float* b1  = (const float*)d_in[3];
  const float* Wmu = (const float*)d_in[4];
  const float* bmu = (const float*)d_in[5];
  const float* Wls = (const float*)d_in[6];
  const float* bls = (const float*)d_in[7];
  float* out = (float*)d_out;

  const int N = in_sizes[0] / CH;   // 100000 (divisible by 16)
  const int E = in_sizes[1] / 2;    // 1600000
  const int* src = ei;
  const int* dst = ei + E;

  // workspace carving (all 256B aligned)
  char* ws = (char*)d_ws;
  size_t off = 0;
  float* dinv = (float*)(ws + off); off += ((size_t)N * 4 + 255) & ~(size_t)255;
  float* bufA = (float*)(ws + off); off += (size_t)N * CH * 4;      // H1, then normalized H
  float* bufB = (float*)(ws + off); off += (size_t)N * CH * 4;      // agg1, then G2
  float* Wcat = (float*)(ws + off); off += (size_t)CH * CH * 4;
  (void)ws_size; (void)n_in; (void)out_size;

  // 1) degrees -> dinv (in place)
  vgcn_deg_init <<<(N + 255) / 256, 256, 0, stream>>>(dinv, N);
  vgcn_deg_accum<<<(E + 255) / 256, 256, 0, stream>>>(dst, dinv, E);
  vgcn_dinv     <<<(N + 255) / 256, 256, 0, stream>>>(dinv, N);

  // 2) layer 1: H1 = X @ W1 ; aggregate ; relu+normalize
  vgcn_gemm128  <<<N / 16, 256, 0, stream>>>(x, W1, bufA, N);
  vgcn_agg_init <<<N, 128, 0, stream>>>(bufA, dinv, b1, bufB);
  vgcn_agg_edges<<<(E + 1) / 2, 256, 0, stream>>>(bufA, src, dst, dinv, bufB, E);
  vgcn_relu_norm<<<N, 128, 0, stream>>>(bufB, bufA);

  // 3) fused layers 2+3: G2 = H @ [Wmu|Wls] ; aggregate straight into d_out
  vgcn_wcat         <<<CH, 128, 0, stream>>>(Wmu, Wls, Wcat);
  vgcn_gemm128      <<<N / 16, 256, 0, stream>>>(bufA, Wcat, bufB, N);
  vgcn_agg_init_out <<<N, 128, 0, stream>>>(bufB, dinv, bmu, bls, out, N);
  vgcn_agg_edges_out<<<(E + 1) / 2, 256, 0, stream>>>(bufB, src, dst, dinv, out, N, E);
}
